// DTD_1941325218069
// MI455X (gfx1250) — matrix operand order repaired and checked by reading, not compiled, not run on hardware
//
#include <hip/hip_runtime.h>

typedef __attribute__((ext_vector_type(16))) _Float16 v16h;
typedef __attribute__((ext_vector_type(8)))  float    v8f;

#define CIN   384
#define C1    256
#define C2    128
#define HW    9216
#define BATCH 2
#define NPIX  (BATCH * HW)     // 18432 total pixels
#define NTILE (HW / 16)        // 576 16-row tiles per batch
#define NCHUNK 4               // column-range split for occupancy
#define JTILES (NTILE / NCHUNK)// 144 column tiles per chunk
#define MASK_THR 0.2f

// Workspace layout (bytes):
//   h       : [C1][NPIX] f32                    18,874,368
//   swzA    : [BATCH*NTILE][32][64] f16          4,718,592 (WMMA A layout)
//   swzB    : [BATCH*NTILE][32][64] f16          4,718,592 (WMMA B layout)
//   simPart : [NCHUNK][NPIX] f32                   294,912
#define WS_H_BYTES    (18874368)
#define WS_SWZA_OFF   (WS_H_BYTES)
#define WS_SWZB_OFF   (WS_SWZA_OFF + 4718592)
#define WS_SMAX_OFF   (WS_SWZB_OFF + 4718592)

// ---------------- Kernel 1: h = relu(W1 @ x + b1), channel-major ----------------
__global__ __launch_bounds__(256) void k1_layer1(
    const float* __restrict__ feats, const float* __restrict__ W1,
    const float* __restrict__ b1, float* __restrict__ h)
{
  int tid   = threadIdx.x;
  int P     = blockIdx.x * 256 + tid;     // global pixel (b*HW + p)
  int obase = blockIdx.y * 32;            // 32 output channels per block
  int b = P / HW, p = P % HW;
  const float* fp = feats + (size_t)b * CIN * HW + p;

  float acc[32];
#pragma unroll
  for (int o = 0; o < 32; ++o) acc[o] = b1[obase + o];

  for (int c = 0; c < CIN; ++c) {
    float x = fp[(size_t)c * HW];         // coalesced across lanes
#pragma unroll
    for (int o = 0; o < 32; ++o)          // W1 index is wave-uniform -> scalar loads
      acc[o] = fmaf(W1[(size_t)(obase + o) * CIN + c], x, acc[o]);
  }
#pragma unroll
  for (int o = 0; o < 32; ++o)
    h[(size_t)(obase + o) * NPIX + P] = fmaxf(acc[o], 0.0f);
}

// ------- Kernel 2: feat = W2 @ h + b2, L2-normalize, emit f16 in WMMA layouts -------
__global__ __launch_bounds__(64) void k2_layer2_norm_swz(
    const float* __restrict__ h, const float* __restrict__ W2,
    const float* __restrict__ b2, _Float16* __restrict__ swzA,
    _Float16* __restrict__ swzB)
{
  __shared__ float featLds[64][129];      // pad 129 -> conflict-free banks
  int tid = threadIdx.x;
  int P   = blockIdx.x * 64 + tid;

  for (int oc = 0; oc < 4; ++oc) {        // 4 chunks of 32 output channels
    float acc[32];
#pragma unroll
    for (int o = 0; o < 32; ++o) acc[o] = b2[oc * 32 + o];
    for (int c = 0; c < C1; ++c) {
      float x = h[(size_t)c * NPIX + P];  // coalesced
#pragma unroll
      for (int o = 0; o < 32; ++o)
        acc[o] = fmaf(W2[(size_t)(oc * 32 + o) * C1 + c], x, acc[o]);
    }
#pragma unroll
    for (int o = 0; o < 32; ++o) featLds[tid][oc * 32 + o] = acc[o];
  }

  float ss = 0.0f;
#pragma unroll
  for (int k = 0; k < C2; ++k) { float v = featLds[tid][k]; ss = fmaf(v, v, ss); }
  float invn = 1.0f / fmaxf(sqrtf(ss), 1e-12f);

  int m = P & 15;                         // row (for A) / col (for B) within 16-tile
  size_t T = (size_t)(P >> 4);            // global tile index (b*NTILE + tile)
  _Float16* baseA = swzA + T * 2048;
  _Float16* baseB = swzB + T * 2048;
#pragma unroll
  for (int k = 0; k < C2; ++k) {
    _Float16 hv = (_Float16)(featLds[tid][k] * invn);
    int c  = k >> 5;                      // 32-wide K chunk
    int kk = k & 31;
    // A-matrix 16x32 f16: lane = m + 16*kk[4]; vgpr = kk[3:1]; half = kk[0]
    int laneA = m + (((kk >> 4) & 1) << 4);
    int idxA  = c * 16 + (((kk & 15) >> 1) << 1) + (kk & 1);
    baseA[laneA * 64 + idxA] = hv;
    // B-matrix 32x16 f16: lane = n + 16*kk[4]; vgpr = kk[3:1]; half = kk[0]
    int laneB = m + (((kk >> 4) & 1) << 4);
    int idxB  = c * 16 + (((kk & 15) >> 1) << 1) + (kk & 1);
    baseB[laneB * 64 + idxB] = hv;
  }
}

// ------- Kernel 3: all-pairs sim via WMMA; async B tiles to LDS; running row max -------
// Each wave owns TWO 16-row A strips; each block covers 256 rows; column range split
// NCHUNK ways across blockIdx.y. Triple-buffered GLOBAL_LOAD_ASYNC_TO_LDS_B128 with
// one barrier per iteration (buffer being overwritten was last read 2 barriers ago).
__global__ __launch_bounds__(256) void k4_sim_nn(
    const _Float16* __restrict__ swzA, const _Float16* __restrict__ swzB,
    float* __restrict__ simPart)
{
  __shared__ __align__(16) _Float16 ldsB[3][2048];   // 3 x 4KB column tiles
  const int tid  = threadIdx.x;
  const int lane = tid & 31;
  const int wave = tid >> 5;
  const int cz   = blockIdx.y;                        // column chunk
  const int b    = blockIdx.z;                        // batch
  const int t0   = blockIdx.x * 16 + wave;            // first row tile of this wave
  const int t1   = t0 + 8;                            // second row tile
  const int g    = lane >> 4;
  const int n    = lane & 15;
  const int jb   = cz * JTILES, je = jb + JTILES;

  // A operands (resident for whole kernel): 2 strips x 4 K-chunks
  v16h a0[4], a1[4];
  {
    const v16h* pa0 = (const v16h*)(swzA + (((size_t)b * NTILE + t0) * 32 + lane) * 64);
    const v16h* pa1 = (const v16h*)(swzA + (((size_t)b * NTILE + t1) * 32 + lane) * 64);
#pragma unroll
    for (int c = 0; c < 4; ++c) { a0[c] = pa0[c]; a1[c] = pa1[c]; }
  }

  float mx0[8], mx1[8];
#pragma unroll
  for (int v = 0; v < 8; ++v) { mx0[v] = -2.0f; mx1[v] = -2.0f; }

  // async copy bookkeeping: one b128 per thread per tile (256 x 16B = 4KB tile)
  const unsigned ldsBase = (unsigned)(size_t)(&ldsB[0][0]) + (unsigned)tid * 16u;
  unsigned long long gaddr = (unsigned long long)(const void*)swzB
                           + ((unsigned long long)b * NTILE + (unsigned long long)jb) * 4096ull
                           + (unsigned long long)tid * 16ull;

  // preload tile jb into buffer 0  (jb % 3 == 0 since JTILES % 3 == 0)
  asm volatile("global_load_async_to_lds_b128 %0, %1, off"
               :: "v"(ldsBase), "v"(gaddr) : "memory");
  gaddr += 4096ull;

  int cur = 0;                        // j % 3
  int nxt = 1;                        // (j+1) % 3
  for (int j = jb; j < je; ++j) {
    if (j + 1 < je) {
      unsigned dst = ldsBase + (unsigned)nxt * 4096u;
      asm volatile("global_load_async_to_lds_b128 %0, %1, off"
                   :: "v"(dst), "v"(gaddr) : "memory");
      gaddr += 4096ull;
      asm volatile("s_wait_asynccnt 0x1" ::: "memory");  // tile j landed (this wave)
    } else {
      asm volatile("s_wait_asynccnt 0x0" ::: "memory");
    }
    __syncthreads();                                     // tile j landed (all waves)

    const v16h* pb = (const v16h*)&ldsB[cur][lane * 64];
    v8f acc0 = {}, acc1 = {};
#pragma unroll
    for (int c = 0; c < 4; ++c) {
      v16h bv = pb[c];                 // shared by both strips
      acc0 = __builtin_amdgcn_wmma_f32_16x16x32_f16(
          false, a0[c], false, bv, (short)0, acc0, false, false);
      acc1 = __builtin_amdgcn_wmma_f32_16x16x32_f16(
          false, a1[c], false, bv, (short)0, acc1, false, false);
    }

    if (j == t0 || j == t1) {          // wave-uniform: diagonal tile, mask self
      int col = j * 16 + n;
#pragma unroll
      for (int v = 0; v < 8; ++v) {
        float s0 = (col == t0 * 16 + v + 8 * g) ? -2.0f : acc0[v];
        float s1 = (col == t1 * 16 + v + 8 * g) ? -2.0f : acc1[v];
        mx0[v] = fmaxf(mx0[v], s0);
        mx1[v] = fmaxf(mx1[v], s1);
      }
    } else {                           // steady state: plain v_max per element
#pragma unroll
      for (int v = 0; v < 8; ++v) {
        mx0[v] = fmaxf(mx0[v], acc0[v]);
        mx1[v] = fmaxf(mx1[v], acc1[v]);
      }
    }

    cur = nxt; nxt = (nxt == 2) ? 0 : nxt + 1;
  }

  // reduce each row's max across the 16 lanes of its N-group
#pragma unroll
  for (int v = 0; v < 8; ++v) {
#pragma unroll
    for (int mask = 1; mask <= 8; mask <<= 1) {
      mx0[v] = fmaxf(mx0[v], __shfl_xor(mx0[v], mask, 32));
      mx1[v] = fmaxf(mx1[v], __shfl_xor(mx1[v], mask, 32));
    }
  }

  if (n == 0) {                        // lanes 0 and 16 write 8 rows per strip
    float* sp = simPart + (size_t)cz * NPIX + (size_t)b * HW;
#pragma unroll
    for (int v = 0; v < 8; ++v) {
      sp[t0 * 16 + v + 8 * g] = mx0[v];
      sp[t1 * 16 + v + 8 * g] = mx1[v];
    }
  }
}

// ------- Kernel 4: combine column-chunk partial maxes -> distance + mask -------
__global__ __launch_bounds__(256) void k5_finalize(
    const float* __restrict__ simPart, float* __restrict__ out)
{
  int P = blockIdx.x * 256 + threadIdx.x;
  float m = simPart[P];
#pragma unroll
  for (int c = 1; c < NCHUNK; ++c) m = fmaxf(m, simPart[(size_t)c * NPIX + P]);
  // unit vectors: ||a-b|| = sqrt(2 - 2*a.b)
  float d = sqrtf(fmaxf(2.0f - 2.0f * m, 0.0f));
  out[P]        = (d > MASK_THR) ? 1.0f : 0.0f;  // mask (B,1,H,W)
  out[NPIX + P] = d;                              // distance (B,N)
}

extern "C" void kernel_launch(void* const* d_in, const int* in_sizes, int n_in,
                              void* d_out, int out_size, void* d_ws, size_t ws_size,
                              hipStream_t stream) {
  const float* feats = (const float*)d_in[0];   // (2,384,96,96)
  const float* W1    = (const float*)d_in[1];   // (256,384)
  const float* b1    = (const float*)d_in[2];   // (256,)
  const float* W2    = (const float*)d_in[3];   // (128,256)
  const float* b2    = (const float*)d_in[4];   // (128,)
  float* out = (float*)d_out;                   // [mask 18432 | distance 18432]

  char* ws = (char*)d_ws;
  float*    h       = (float*)ws;
  _Float16* swzA    = (_Float16*)(ws + WS_SWZA_OFF);
  _Float16* swzB    = (_Float16*)(ws + WS_SWZB_OFF);
  float*    simPart = (float*)(ws + WS_SMAX_OFF);

  k1_layer1<<<dim3(NPIX / 256, 8), 256, 0, stream>>>(feats, W1, b1, h);
  k2_layer2_norm_swz<<<dim3(NPIX / 64), 64, 0, stream>>>(h, W2, b2, swzA, swzB);
  k4_sim_nn<<<dim3(NTILE / 16 /*36*/, NCHUNK, BATCH), 256, 0, stream>>>(swzA, swzB, simPart);
  k5_finalize<<<dim3(NPIX / 256), 256, 0, stream>>>(simPart, out);
}
